// GRUWithAdditiveAttentionDecoder_77103252898323
// MI455X (gfx1250) — matrix operand order repaired and checked by reading, not compiled, not run on hardware
//
#include <hip/hip_runtime.h>
#include <hip/hip_bf16.h>
#include <math.h>

#define V_  32000
#define E_  256
#define H_  256
#define B_  32
#define T_  256
#define L_  512   // 2T fixed KV length
#define G3_ 768   // 3H gate width

typedef __attribute__((ext_vector_type(16))) _Float16 v16h;
typedef __attribute__((ext_vector_type(8)))  _Float16 v8h;
typedef __attribute__((ext_vector_type(8)))  float    v8f;

#ifndef __has_builtin
#define __has_builtin(x) 0
#endif
#if __has_builtin(__builtin_amdgcn_global_load_async_to_lds_b128)
#define HAVE_ASYNC_LDS 1
typedef int v4i_b128 __attribute__((vector_size(16)));
typedef v4i_b128 __attribute__((address_space(1)))* agptr_t;   // global (AS1) int4*
typedef v4i_b128 __attribute__((address_space(3)))* alptr_t;   // LDS (AS3) int4*
#else
#define HAVE_ASYNC_LDS 0
#endif

// ---------------------------------------------------------------- WMMA helpers
static __device__ __forceinline__ v8f wmma_16x16x32(v16h a, v16h b, v8f c) {
  // D = A(16x32 f16) * B(32x16 f16) + C(16x16 f32)
  return __builtin_amdgcn_wmma_f32_16x16x32_f16(false, a, false, b, (short)0, c,
                                                false, false);
}

// A fragment: row-major A[.. , lda] f16, 16x32 tile at (m0, k0).
// ISA 7.12.2 (16-bit A 16x32): lane m = lane%16; half = lane/16;
// VGPR i in 0..3 -> K = 8*half + 2i + {0,1}; VGPR i in 4..7 -> K = 16 + 8*half + 2(i-4) + {0,1}
static __device__ __forceinline__ v16h load_frag_a(const _Float16* A, int lda,
                                                   int m0, int k0) {
  const int lane = (int)(threadIdx.x & 31u);
  const int half = lane >> 4;
  const _Float16* p = A + (size_t)(m0 + (lane & 15)) * (size_t)lda + k0;
  v16h r;
#pragma unroll
  for (int i = 0; i < 4; ++i) {
    r[2 * i + 0]     = p[8 * half + 2 * i + 0];
    r[2 * i + 1]     = p[8 * half + 2 * i + 1];
    r[8 + 2 * i + 0] = p[16 + 8 * half + 2 * i + 0];
    r[8 + 2 * i + 1] = p[16 + 8 * half + 2 * i + 1];
  }
  return r;
}

// B fragment for X @ W^T with W row-major [N][ldw]: B(k,n) = W[n][k].
// lane n = lane%16, half = lane/16: elems 0..7 -> K = k0+8*half+i ; elems 8..15 -> K = k0+16+8*half+i
static __device__ __forceinline__ v16h load_frag_b_wt(const _Float16* W, int ldw,
                                                      int n0, int k0) {
  const int lane = (int)(threadIdx.x & 31u);
  const _Float16* p = W + (size_t)(n0 + (lane & 15)) * (size_t)ldw + k0 + 8 * (lane >> 4);
  v16h r;
#pragma unroll
  for (int i = 0; i < 8; ++i) { r[i] = p[i]; r[8 + i] = p[16 + i]; }
  return r;
}

// C/D store: lane n = n0 + lane%16; rows m = m0 + 8*(lane/16) + i
static __device__ __forceinline__ void store_frag_f32(float* C, int ldc, int m0,
                                                      int n0, v8f c) {
  const int lane = (int)(threadIdx.x & 31u);
  const int n = n0 + (lane & 15);
  const int mb = m0 + 8 * (lane >> 4);
#pragma unroll
  for (int i = 0; i < 8; ++i) C[(size_t)(mb + i) * (size_t)ldc + n] = c[i];
}

// ---------------------------------------------------------------- prep kernels
__global__ void cvt_f32_f16_kernel(const float* __restrict__ src,
                                   _Float16* __restrict__ dst, size_t n) {
  for (size_t i = (size_t)blockIdx.x * blockDim.x + threadIdx.x; i < n;
       i += (size_t)gridDim.x * blockDim.x)
    dst[i] = (_Float16)src[i];
}

__global__ void zero_f16_kernel(_Float16* __restrict__ dst, size_t n) {
  for (size_t i = (size_t)blockIdx.x * blockDim.x + threadIdx.x; i < n;
       i += (size_t)gridDim.x * blockDim.x)
    dst[i] = (_Float16)0.0f;
}

// emb16[t][b][e] = embedding[x[b][t]][e]
__global__ void gather_emb_kernel(const int* __restrict__ x,
                                  const float* __restrict__ emb,
                                  _Float16* __restrict__ out) {
  const size_t n = (size_t)T_ * B_ * E_;
  for (size_t i = (size_t)blockIdx.x * blockDim.x + threadIdx.x; i < n;
       i += (size_t)gridDim.x * blockDim.x) {
    int e = (int)(i & 255u);
    int b = (int)((i >> 8) & 31u);
    int t = (int)(i >> 13);
    out[i] = (_Float16)emb[(size_t)x[b * T_ + t] * E_ + e];
  }
}

// h0[layer][b][h] = st[2*layer][b][h] + st[2*layer+1][b][h]
__global__ void init_h_kernel(const float* __restrict__ st,
                              float* __restrict__ h32,
                              _Float16* __restrict__ h16) {
  const int n = 2 * B_ * H_;
  for (int i = blockIdx.x * blockDim.x + threadIdx.x; i < n;
       i += gridDim.x * blockDim.x) {
    int layer = i >> 13;           // / (B*H)
    int rem = i & 8191;
    float v = st[(size_t)(2 * layer) * (B_ * H_) + rem] +
              st[(size_t)(2 * layer + 1) * (B_ * H_) + rem];
    h32[i] = v;
    h16[i] = (_Float16)v;
  }
}

// ---------------------------------------------------------------- prologue GEMMs
// kv16[b][s][n] = sum_d enc16[s][b][d] * encw16[n][d] + encb[n],  s < T+1
// A rows r = s*B + b (enc is (T+1,B,2H) contiguous), M = 8224 = 514*16, K = 512
__global__ void encproj_gemm_kernel(const _Float16* __restrict__ enc16,
                                    const _Float16* __restrict__ encw16,
                                    const float* __restrict__ encb,
                                    _Float16* __restrict__ kv16) {
  const int wave = (int)(threadIdx.x >> 5);
  const int job = blockIdx.x * 8 + wave;       // 8224 jobs
  const int m0 = (job >> 4) * 16;
  const int n0 = (job & 15) * 16;
  v8f c = {};
  for (int k0 = 0; k0 < 512; k0 += 32)
    c = wmma_16x16x32(load_frag_a(enc16, 512, m0, k0),
                      load_frag_b_wt(encw16, 512, n0, k0), c);
  const int lane = (int)(threadIdx.x & 31u);
  const int n = n0 + (lane & 15);
  const int mb = m0 + 8 * (lane >> 4);
  const float bv = encb[n];
#pragma unroll
  for (int i = 0; i < 8; ++i) {
    int r = mb + i;
    int s = r >> 5;              // / B
    int b = r & 31;
    kv16[((size_t)b * L_ + s) * H_ + n] = (_Float16)(c[i] + bv);
  }
}

// kproj16[r][n] = sum_h kv16[r][h] * wk16[n][h], r = b*L+l, M = 16384, K = 256
__global__ void kproj0_gemm_kernel(const _Float16* __restrict__ kv16,
                                   const _Float16* __restrict__ wk16,
                                   _Float16* __restrict__ kproj16) {
  const int job = blockIdx.x * 8 + (int)(threadIdx.x >> 5);   // 16384 jobs
  const int m0 = (job >> 4) * 16;
  const int n0 = (job & 15) * 16;
  v8f c = {};
  for (int k0 = 0; k0 < 256; k0 += 32)
    c = wmma_16x16x32(load_frag_a(kv16, 256, m0, k0),
                      load_frag_b_wt(wk16, 256, n0, k0), c);
  const int lane = (int)(threadIdx.x & 31u);
  const int n = n0 + (lane & 15);
  const int mb = m0 + 8 * (lane >> 4);
#pragma unroll
  for (int i = 0; i < 8; ++i)
    kproj16[(size_t)(mb + i) * H_ + n] = (_Float16)c[i];
}

// ---------------------------------------------------------------- per-step attention
// one block per batch row b; 256 threads
__global__ void attn_step_kernel(const float* __restrict__ h32,
                                 const float* __restrict__ wq,
                                 const float* __restrict__ wv,
                                 const _Float16* __restrict__ kproj16,
                                 const _Float16* __restrict__ kv16,
                                 const _Float16* __restrict__ emb16,
                                 _Float16* __restrict__ xin16,
                                 float* __restrict__ attn_out, int t) {
  const int b = blockIdx.x;
  const int tid = (int)threadIdx.x;       // 0..255
  __shared__ float qs[H_];
  __shared__ float wvs[H_];
  __shared__ float sc[L_];
  __shared__ float red[256];

  // q = h1 @ wq^T  (h[-1] is layer-1 hidden state)
  const float* h1 = h32 + (size_t)(B_ + b) * H_;
  {
    const float* wqr = wq + (size_t)tid * H_;
    float acc = 0.f;
    for (int k = 0; k < H_; ++k) acc += h1[k] * wqr[k];
    qs[tid] = acc;
    wvs[tid] = wv[tid];
  }
  __syncthreads();

  const int valid = T_ + 1 + t;           // 257..512
  for (int l = tid; l < L_; l += 256) {
    float s;
    if (l < valid) {
      // kproj rows are 512-byte aligned: use 16B vector loads
      const v8h* kp8 = (const v8h*)(kproj16 + ((size_t)b * L_ + l) * H_);
      float sum = 0.f;
      for (int h8 = 0; h8 < H_ / 8; ++h8) {
        const v8h kvv = kp8[h8];
#pragma unroll
        for (int j = 0; j < 8; ++j) {
          const int h = h8 * 8 + j;
          sum += tanhf((float)kvv[j] + qs[h]) * wvs[h];
        }
      }
      s = sum;
    } else {
      s = -1e30f;
    }
    sc[l] = s;
  }
  __syncthreads();

  // softmax over sc[0..L)
  red[tid] = fmaxf(sc[tid], sc[tid + 256]);
  __syncthreads();
  for (int s = 128; s > 0; s >>= 1) {
    if (tid < s) red[tid] = fmaxf(red[tid], red[tid + s]);
    __syncthreads();
  }
  const float mx = red[0];
  __syncthreads();

  const int l0 = tid, l1 = tid + 256;
  const float e0 = (l0 < valid) ? expf(sc[l0] - mx) : 0.f;
  const float e1 = (l1 < valid) ? expf(sc[l1] - mx) : 0.f;
  red[tid] = e0 + e1;
  __syncthreads();
  for (int s = 128; s > 0; s >>= 1) {
    if (tid < s) red[tid] += red[tid + s];
    __syncthreads();
  }
  const float inv = 1.0f / red[0];
  const float a0 = e0 * inv, a1 = e1 * inv;
  sc[l0] = a0;
  sc[l1] = a1;
  float* arow = attn_out + ((size_t)b * T_ + t) * L_;
  arow[l0] = a0;
  arow[l1] = a1;
  __syncthreads();

  // context[n] = sum_l attn[l] * kv[b][l][n];  xin = [context | emb_t]
  {
    const int n = tid;
    const _Float16* kvb = kv16 + (size_t)b * L_ * H_ + n;
    float ctx = 0.f;
    for (int l = 0; l < valid; ++l)
      ctx += sc[l] * (float)kvb[(size_t)l * H_];
    xin16[(size_t)b * 512 + n] = (_Float16)ctx;
    xin16[(size_t)b * 512 + 256 + n] = emb16[((size_t)t * B_ + b) * E_ + n];
  }
}

// ---------------------------------------------------------------- per-step GRU (1 block, 32 waves)
__global__ void __launch_bounds__(1024)
gru_step_kernel(const _Float16* __restrict__ xin16,
                const _Float16* __restrict__ wih0, const _Float16* __restrict__ whh0,
                const _Float16* __restrict__ wih1, const _Float16* __restrict__ whh1,
                const float* __restrict__ bih0, const float* __restrict__ bhh0,
                const float* __restrict__ bih1, const float* __restrict__ bhh1,
                float* __restrict__ h32, _Float16* __restrict__ h16,
                float* __restrict__ gi0, float* __restrict__ gh0,
                float* __restrict__ gi1, float* __restrict__ gh1,
                _Float16* __restrict__ outs16, _Float16* __restrict__ kv16,
                _Float16* __restrict__ kproj16, const _Float16* __restrict__ wk16,
                int t) {
  const int tid = (int)threadIdx.x;
  const int wave = tid >> 5;
  const int pos0 = T_ + 1 + t;
  const int pos = (pos0 < L_) ? pos0 : (L_ - 1);  // jax dynamic_update_slice clamps

  // phase 1: gi0 = xin(32x512) @ wih0^T(768x512); gh0 = h0(32x256) @ whh0^T(768x256)
  for (int job = wave; job < 96; job += 32) {
    const int m0 = (job / 48) * 16;
    const int n0 = (job % 48) * 16;
    v8f c = {};
    for (int k0 = 0; k0 < 512; k0 += 32)
      c = wmma_16x16x32(load_frag_a(xin16, 512, m0, k0),
                        load_frag_b_wt(wih0, 512, n0, k0), c);
    store_frag_f32(gi0, G3_, m0, n0, c);
    v8f c2 = {};
    for (int k0 = 0; k0 < 256; k0 += 32)
      c2 = wmma_16x16x32(load_frag_a(h16, 256, m0, k0),
                         load_frag_b_wt(whh0, 256, n0, k0), c2);
    store_frag_f32(gh0, G3_, m0, n0, c2);
  }
  __threadfence_block();
  __syncthreads();

  // phase 2: layer-0 gates
  for (int idx = tid; idx < B_ * H_; idx += 1024) {
    const int b = idx >> 8, n = idx & 255;
    const float ir = gi0[(size_t)b * G3_ + n]        + bih0[n];
    const float iz = gi0[(size_t)b * G3_ + 256 + n]  + bih0[256 + n];
    const float in_ = gi0[(size_t)b * G3_ + 512 + n] + bih0[512 + n];
    const float hr = gh0[(size_t)b * G3_ + n]        + bhh0[n];
    const float hz = gh0[(size_t)b * G3_ + 256 + n]  + bhh0[256 + n];
    const float hn = gh0[(size_t)b * G3_ + 512 + n]  + bhh0[512 + n];
    const float r = 1.f / (1.f + expf(-(ir + hr)));
    const float z = 1.f / (1.f + expf(-(iz + hz)));
    const float nn = tanhf(in_ + r * hn);
    const float hnew = (1.f - z) * nn + z * h32[idx];
    h32[idx] = hnew;
    h16[idx] = (_Float16)hnew;
  }
  __threadfence_block();
  __syncthreads();

  // phase 3: gi1 = h0new @ wih1^T; gh1 = h1old @ whh1^T  (both K=256)
  for (int job = wave; job < 96; job += 32) {
    const int m0 = (job / 48) * 16;
    const int n0 = (job % 48) * 16;
    v8f c = {};
    v8f c2 = {};
    for (int k0 = 0; k0 < 256; k0 += 32) {
      c = wmma_16x16x32(load_frag_a(h16, 256, m0, k0),
                        load_frag_b_wt(wih1, 256, n0, k0), c);
      c2 = wmma_16x16x32(load_frag_a(h16 + (size_t)B_ * H_, 256, m0, k0),
                         load_frag_b_wt(whh1, 256, n0, k0), c2);
    }
    store_frag_f32(gi1, G3_, m0, n0, c);
    store_frag_f32(gh1, G3_, m0, n0, c2);
  }
  __threadfence_block();
  __syncthreads();

  // phase 4: layer-1 gates, emit output / kv append
  for (int idx = tid; idx < B_ * H_; idx += 1024) {
    const int b = idx >> 8, n = idx & 255;
    const float ir = gi1[(size_t)b * G3_ + n]        + bih1[n];
    const float iz = gi1[(size_t)b * G3_ + 256 + n]  + bih1[256 + n];
    const float in_ = gi1[(size_t)b * G3_ + 512 + n] + bih1[512 + n];
    const float hr = gh1[(size_t)b * G3_ + n]        + bhh1[n];
    const float hz = gh1[(size_t)b * G3_ + 256 + n]  + bhh1[256 + n];
    const float hn = gh1[(size_t)b * G3_ + 512 + n]  + bhh1[512 + n];
    const float r = 1.f / (1.f + expf(-(ir + hr)));
    const float z = 1.f / (1.f + expf(-(iz + hz)));
    const float nn = tanhf(in_ + r * hn);
    const float hnew = (1.f - z) * nn + z * h32[B_ * H_ + idx];
    h32[B_ * H_ + idx] = hnew;
    const _Float16 hh = (_Float16)hnew;
    h16[B_ * H_ + idx] = hh;
    outs16[((size_t)b * T_ + t) * H_ + n] = hh;
    kv16[((size_t)b * L_ + pos) * H_ + n] = hh;
  }
  __threadfence_block();
  __syncthreads();

  // phase 5: kproj append = h1new(32x256) @ wk^T(256x256) -> 32 tiles, 1 per wave
  {
    const int m0 = (wave >> 4) * 16;
    const int n0 = (wave & 15) * 16;
    v8f c = {};
    for (int k0 = 0; k0 < 256; k0 += 32)
      c = wmma_16x16x32(load_frag_a(h16 + (size_t)B_ * H_, 256, m0, k0),
                        load_frag_b_wt(wk16, 256, n0, k0), c);
    const int lane = (int)(threadIdx.x & 31u);
    const int n = n0 + (lane & 15);
    const int mb = m0 + 8 * (lane >> 4);
#pragma unroll
    for (int i = 0; i < 8; ++i) {
      const int b = mb + i;
      kproj16[((size_t)b * L_ + pos) * H_ + n] = (_Float16)c[i];
    }
  }
}

// ---------------------------------------------------------------- logits GEMM
// One block = one 16-row m-tile staged in LDS (async DMA); 8 waves x 64 cols = 512 cols/block.
#define LOGITS_CG 63   // ceil(32000 / 512)
__global__ void __launch_bounds__(256)
logits_gemm_kernel(const _Float16* __restrict__ A,
                   const _Float16* __restrict__ W,
                   const float* __restrict__ bias,
                   float* __restrict__ out) {
  __shared__ _Float16 atile[16 * 256];   // 8 KB
  const int tid = (int)threadIdx.x;
  const int wave = tid >> 5;
  const int mt = blockIdx.x / LOGITS_CG;
  const int cg = blockIdx.x % LOGITS_CG;
  const int m0 = mt * 16;

  // stage the A tile (16x256 f16): each thread moves 16 f16 (two b128 ops)
  {
    const _Float16* gsrc = A + (size_t)m0 * 256 + (size_t)tid * 16;
    _Float16* ldst = atile + tid * 16;
#if HAVE_ASYNC_LDS
    __builtin_amdgcn_global_load_async_to_lds_b128((agptr_t)gsrc, (alptr_t)ldst, 0, 0);
    __builtin_amdgcn_global_load_async_to_lds_b128((agptr_t)(gsrc + 8), (alptr_t)(ldst + 8), 0, 0);
    asm volatile("s_wait_asynccnt 0x0" ::: "memory");
#else
    const uint4* g4 = (const uint4*)gsrc;
    uint4* l4 = (uint4*)ldst;
    l4[0] = g4[0];
    l4[1] = g4[1];
#endif
  }
  __syncthreads();

  const int n0 = cg * 512 + wave * 64;   // wave-uniform
  if (n0 < V_) {
    v8f c[4] = {{}, {}, {}, {}};
    for (int k0 = 0; k0 < 256; k0 += 32) {
      const v16h a = load_frag_a(atile, 256, 0, k0);   // ds_load from LDS
      __builtin_prefetch(W + (size_t)n0 * 256 + k0 + 64, 0, 1);
#pragma unroll
      for (int j = 0; j < 4; ++j)
        c[j] = wmma_16x16x32(a, load_frag_b_wt(W, 256, n0 + 16 * j, k0), c[j]);
    }
    const int lane = (int)(threadIdx.x & 31u);
    const int nl = lane & 15;
    const int mb = m0 + 8 * (lane >> 4);
#pragma unroll
    for (int j = 0; j < 4; ++j) {
      const int n = n0 + 16 * j + nl;
      const float bv = bias[n];
#pragma unroll
      for (int i = 0; i < 8; ++i)
        out[(size_t)(mb + i) * V_ + n] = c[j][i] + bv;
    }
  }
}

// ---------------------------------------------------------------- launch
extern "C" void kernel_launch(void* const* d_in, const int* in_sizes, int n_in,
                              void* d_out, int out_size, void* d_ws, size_t ws_size,
                              hipStream_t stream) {
  (void)in_sizes; (void)n_in; (void)out_size; (void)ws_size;
  const int*   x     = (const int*)d_in[0];
  const float* enc   = (const float*)d_in[1];
  const float* encfs = (const float*)d_in[2];
  const float* embed = (const float*)d_in[3];
  const float* encw  = (const float*)d_in[4];
  const float* encb  = (const float*)d_in[5];
  const float* wq    = (const float*)d_in[6];
  const float* wk    = (const float*)d_in[7];
  const float* wv    = (const float*)d_in[8];
  const float* wih0  = (const float*)d_in[9];
  const float* whh0  = (const float*)d_in[10];
  const float* bih0  = (const float*)d_in[11];
  const float* bhh0  = (const float*)d_in[12];
  const float* wih1  = (const float*)d_in[13];
  const float* whh1  = (const float*)d_in[14];
  const float* bih1  = (const float*)d_in[15];
  const float* bhh1  = (const float*)d_in[16];
  const float* dw    = (const float*)d_in[17];
  const float* db    = (const float*)d_in[18];

  char* ws = (char*)d_ws;
  size_t off = 0;
  auto a16 = [&](size_t n) -> _Float16* {
    _Float16* p = (_Float16*)(ws + off);
    off = (off + n * sizeof(_Float16) + 255) & ~(size_t)255;
    return p;
  };
  auto a32 = [&](size_t n) -> float* {
    float* p = (float*)(ws + off);
    off = (off + n * sizeof(float) + 255) & ~(size_t)255;
    return p;
  };

  _Float16* dw16    = a16((size_t)V_ * H_);
  _Float16* enc16   = a16((size_t)(T_ + 1) * B_ * (2 * H_));
  _Float16* wih0_16 = a16((size_t)G3_ * 512);
  _Float16* whh0_16 = a16((size_t)G3_ * H_);
  _Float16* wih1_16 = a16((size_t)G3_ * H_);
  _Float16* whh1_16 = a16((size_t)G3_ * H_);
  _Float16* wk16    = a16((size_t)H_ * H_);
  _Float16* encw16  = a16((size_t)H_ * 2 * H_);
  _Float16* emb16   = a16((size_t)T_ * B_ * E_);
  _Float16* kv16    = a16((size_t)B_ * L_ * H_);
  _Float16* kproj16 = a16((size_t)B_ * L_ * H_);
  _Float16* h16     = a16((size_t)2 * B_ * H_);
  _Float16* xin16   = a16((size_t)B_ * 512);
  _Float16* outs16  = a16((size_t)B_ * T_ * H_);
  float*    h32     = a32((size_t)2 * B_ * H_);
  float*    gi0     = a32((size_t)B_ * G3_);
  float*    gh0     = a32((size_t)B_ * G3_);
  float*    gi1     = a32((size_t)B_ * G3_);
  float*    gh1     = a32((size_t)B_ * G3_);

  float* logits   = (float*)d_out;
  float* attn_out = (float*)d_out + (size_t)B_ * T_ * V_;

  // ---- prologue: conversions / gathers / zero-fills
  cvt_f32_f16_kernel<<<2048, 256, 0, stream>>>(dw, dw16, (size_t)V_ * H_);
  cvt_f32_f16_kernel<<<2048, 256, 0, stream>>>(enc, enc16, (size_t)(T_ + 1) * B_ * 2 * H_);
  cvt_f32_f16_kernel<<<512, 256, 0, stream>>>(wih0, wih0_16, (size_t)G3_ * 512);
  cvt_f32_f16_kernel<<<512, 256, 0, stream>>>(whh0, whh0_16, (size_t)G3_ * H_);
  cvt_f32_f16_kernel<<<512, 256, 0, stream>>>(wih1, wih1_16, (size_t)G3_ * H_);
  cvt_f32_f16_kernel<<<512, 256, 0, stream>>>(whh1, whh1_16, (size_t)G3_ * H_);
  cvt_f32_f16_kernel<<<256, 256, 0, stream>>>(wk, wk16, (size_t)H_ * H_);
  cvt_f32_f16_kernel<<<256, 256, 0, stream>>>(encw, encw16, (size_t)H_ * 2 * H_);
  gather_emb_kernel<<<2048, 256, 0, stream>>>(x, embed, emb16);
  init_h_kernel<<<64, 256, 0, stream>>>(encfs, h32, h16);
  zero_f16_kernel<<<2048, 256, 0, stream>>>(kv16, (size_t)B_ * L_ * H_);
  zero_f16_kernel<<<2048, 256, 0, stream>>>(kproj16, (size_t)B_ * L_ * H_);

  // ---- encoder projection + key pre-projection (WMMA)
  encproj_gemm_kernel<<<1028, 256, 0, stream>>>(enc16, encw16, encb, kv16);
  kproj0_gemm_kernel<<<2048, 256, 0, stream>>>(kv16, wk16, kproj16);

  // ---- sequential decode
  for (int t = 0; t < T_; ++t) {
    attn_step_kernel<<<B_, 256, 0, stream>>>(h32, wq, wv, kproj16, kv16, emb16,
                                             xin16, attn_out, t);
    gru_step_kernel<<<1, 1024, 0, stream>>>(xin16, wih0_16, whh0_16, wih1_16,
                                            whh1_16, bih0, bhh0, bih1, bhh1,
                                            h32, h16, gi0, gh0, gi1, gh1,
                                            outs16, kv16, kproj16, wk16, t);
  }

  // ---- final vocabulary projection (dominant GEMM, WMMA f16, LDS-staged A)
  logits_gemm_kernel<<<512 * LOGITS_CG, 256, 0, stream>>>(outs16, dw16, db, logits);
}
